// HardNegCLIPInfoNCE_Pro_NoLabel_44890998178046
// MI455X (gfx1250) — matrix operand order repaired and checked by reading, not compiled, not run on hardware
//
#include <hip/hip_runtime.h>

// ---------------- problem constants ----------------
#define BSZ   8192          // batch
#define DSZ   1024          // embed dim
#define KHARD 32
#define KRAND 32

// ---------------- GEMM tiling ----------------
#define BAND    64          // rows owned by one workgroup (4 m-tiles)
#define CHUNK   128         // columns processed per iteration (8 n-tiles)
#define NSUB    4           // n-subtiles (16 cols each) per wave
#define KSTEP   32          // K per WMMA
#define APAD    8           // u16 pad per LDS A row (16B) -> kills bank conflicts
#define ASTRIDE (DSZ + APAD)
#define SSTRIDE (CHUNK + 4) // f32 pad for the staging tile

typedef __attribute__((ext_vector_type(16))) __bf16 v16bf;
typedef __attribute__((ext_vector_type(8)))  float  v8f;

union ABfrag { v16bf v; unsigned short u[16]; };

__device__ __forceinline__ unsigned short f32_to_bf16(float f) {
  unsigned int u = __float_as_uint(f);
  u += 0x7FFFu + ((u >> 16) & 1u);        // round-to-nearest-even
  return (unsigned short)(u >> 16);
}
__device__ __forceinline__ float bflo(unsigned int u) { return __uint_as_float(u << 16); }
__device__ __forceinline__ float bfhi(unsigned int u) { return __uint_as_float(u & 0xFFFF0000u); }

__device__ __forceinline__ float dot8(uint4 a, uint4 b) {
  float s = bflo(a.x) * bflo(b.x) + bfhi(a.x) * bfhi(b.x);
  s += bflo(a.y) * bflo(b.y) + bfhi(a.y) * bfhi(b.y);
  s += bflo(a.z) * bflo(b.z) + bfhi(a.z) * bfhi(b.z);
  s += bflo(a.w) * bflo(b.w) + bfhi(a.w) * bfhi(b.w);
  return s;
}

// 8 bf16 from p0 + 8 bf16 from p1 (two 16B vectors) -> fragment halves
__device__ __forceinline__ void load_frag(ABfrag& f, const unsigned short* p0,
                                          const unsigned short* p1) {
  *(uint4*)&f.u[0] = *(const uint4*)p0;
  *(uint4*)&f.u[8] = *(const uint4*)p1;
}

// ------------------------------------------------------------------
// 1) normalize rows of both matrices, emit bf16
// ------------------------------------------------------------------
__global__ __launch_bounds__(256) void normalize_kernel(
    const float* __restrict__ img, const float* __restrict__ txt,
    unsigned short* __restrict__ Abf, unsigned short* __restrict__ Bbf) {
  const int b   = blockIdx.x;
  const int tid = threadIdx.x;
  const float*    src = (b < BSZ) ? img + (size_t)b * DSZ : txt + (size_t)(b - BSZ) * DSZ;
  unsigned short* dst = (b < BSZ) ? Abf + (size_t)b * DSZ : Bbf + (size_t)(b - BSZ) * DSZ;

  __shared__ float red[256];
  float ss = 0.f;
  for (int k = tid; k < DSZ; k += 256) { float v = src[k]; ss += v * v; }
  red[tid] = ss;
  __syncthreads();
  for (int s = 128; s > 0; s >>= 1) {
    if (tid < s) red[tid] += red[tid + s];
    __syncthreads();
  }
  const float rn = 1.0f / fmaxf(sqrtf(red[0]), 1e-12f);
  for (int k = tid; k < DSZ; k += 256) dst[k] = f32_to_bf16(src[k] * rn);
}

__global__ void zero_kernel(float* out) { out[0] = 0.0f; }

// ------------------------------------------------------------------
// 2) fused bf16-WMMA GEMM band + diag mask + streaming per-row top-32
//    grid: (BSZ/BAND, 2); 256 threads = 8 waves
//    wave w: m-tile = w>>1 (16 rows), n-group = w&1 (64 cols = 4 subtiles)
// ------------------------------------------------------------------
__global__ __launch_bounds__(256) void sim_topk_kernel(
    const unsigned short* __restrict__ Abf,
    const unsigned short* __restrict__ Bbf,
    float* __restrict__ posOut,   // [2*BSZ]
    float* __restrict__ hardV,    // [2*BSZ*KHARD]
    int*   __restrict__ hardI) {  // [2*BSZ*KHARD]
  extern __shared__ char smem[];
  unsigned short* sA    = (unsigned short*)smem;                       // BAND*ASTRIDE u16
  float*          sTile = (float*)(smem + (size_t)BAND * ASTRIDE * 2); // BAND*SSTRIDE f32
  float*          sV    = sTile + BAND * SSTRIDE;                      // BAND*KHARD
  int*            sI    = (int*)(sV + BAND * KHARD);                   // BAND*KHARD
  float*          sPos  = (float*)(sI + BAND * KHARD);                 // BAND

  const int dir  = blockIdx.y;
  const unsigned short* __restrict__ Am = dir ? Bbf : Abf;   // row matrix
  const unsigned short* __restrict__ Bm = dir ? Abf : Bbf;   // col matrix
  const int row0 = blockIdx.x * BAND;
  const int tid  = threadIdx.x;
  const int lane = tid & 31;
  const int wave = tid >> 5;

  // ---- stage A band into LDS (16B vectors) ----
  for (int idx = tid; idx < BAND * (DSZ / 8); idx += 256) {
    const int r = idx / (DSZ / 8);
    const int c = (idx % (DSZ / 8)) * 8;
    *(uint4*)(sA + (size_t)r * ASTRIDE + c) =
        *(const uint4*)(Am + (size_t)(row0 + r) * DSZ + c);
  }
  for (int idx = tid; idx < BAND * KHARD; idx += 256) { sV[idx] = -3.0e30f; sI[idx] = -1; }
  __syncthreads();

  const int mBase  = (wave >> 1) * 16;    // 0..48
  const int nGrp   = (wave & 1) * 64;     // 0 or 64
  const int l15    = lane & 15;
  const int half   = lane >> 4;
  const int half8  = half * 8;
  const unsigned short* aptr = sA + (size_t)(mBase + l15) * ASTRIDE;

  for (int chunk = 0; chunk < BSZ / CHUNK; ++chunk) {
    const int col0 = chunk * CHUNK;

    const unsigned short* bp[NSUB];
#pragma unroll
    for (int t = 0; t < NSUB; ++t)
      bp[t] = Bm + (size_t)(col0 + nGrp + t * 16 + l15) * DSZ + half * 8;
    if (chunk + 1 < BSZ / CHUNK)
      __builtin_prefetch(bp[0] + (size_t)CHUNK * DSZ, 0, 1);  // next chunk -> global_prefetch

    v8f acc[NSUB] = {};
    ABfrag aC, aN, bC[NSUB], bN[NSUB];

    // prologue: K-step 0
    load_frag(aC, aptr + half8, aptr + half8 + 16);
#pragma unroll
    for (int t = 0; t < NSUB; ++t) load_frag(bC[t], bp[t], bp[t] + 16);

    for (int k = 0; k < DSZ; k += 2 * KSTEP) {
      const int k1 = k + KSTEP;
      // stage k+32 while computing on k
      load_frag(aN, aptr + k1 + half8, aptr + k1 + half8 + 16);
#pragma unroll
      for (int t = 0; t < NSUB; ++t) load_frag(bN[t], bp[t] + k1, bp[t] + k1 + 16);
#pragma unroll
      for (int t = 0; t < NSUB; ++t)
        acc[t] = __builtin_amdgcn_wmma_f32_16x16x32_bf16(false, aC.v, false, bC[t].v,
                                                         (short)0, acc[t], false, false);
      const int k2 = k + 2 * KSTEP;
      if (k2 < DSZ) {   // stage k+64 while computing on k+32
        load_frag(aC, aptr + k2 + half8, aptr + k2 + half8 + 16);
#pragma unroll
        for (int t = 0; t < NSUB; ++t) load_frag(bC[t], bp[t] + k2, bp[t] + k2 + 16);
      }
#pragma unroll
      for (int t = 0; t < NSUB; ++t)
        acc[t] = __builtin_amdgcn_wmma_f32_16x16x32_bf16(false, aN.v, false, bN[t].v,
                                                         (short)0, acc[t], false, false);
    }

    __syncthreads();                       // top-k readers of previous tile done
    // C layout: VGPR r -> row r + 8*half, col = lane&15
#pragma unroll
    for (int t = 0; t < NSUB; ++t) {
      const int colT = nGrp + t * 16 + l15;
#pragma unroll
      for (int r = 0; r < 8; ++r)
        sTile[(size_t)(mBase + r + 8 * half) * SSTRIDE + colT] = acc[t][r];
    }
    __syncthreads();

    // ---- streaming top-32 update, one thread per band row ----
    if (tid < BAND) {
      const int grow = row0 + tid;
      float* vv = sV + tid * KHARD;
      int*   ii = sI + tid * KHARD;
      float  thr = vv[KHARD - 1];
      for (int c = 0; c < CHUNK; ++c) {
        const float v = sTile[tid * SSTRIDE + c];
        const int gcol = col0 + c;
        if (gcol == grow) { sPos[tid] = v; continue; }   // diag -> positive, never a negative
        if (v > thr) {
          int p = KHARD - 1;
          while (p > 0 && vv[p - 1] < v) { vv[p] = vv[p - 1]; ii[p] = ii[p - 1]; --p; }
          vv[p] = v; ii[p] = gcol;
          thr = vv[KHARD - 1];
        }
      }
    }
  }
  __syncthreads();

  if (tid < BAND) {
    const size_t base = (size_t)dir * BSZ + (row0 + tid);
    posOut[base] = sPos[tid];
    for (int j = 0; j < KHARD; ++j) {
      hardV[base * KHARD + j] = sV[tid * KHARD + j];
      hardI[base * KHARD + j] = sI[tid * KHARD + j];
    }
  }
}

// ------------------------------------------------------------------
// 3) random negatives + InfoNCE row loss; grid (BSZ, 2), 128 threads
// ------------------------------------------------------------------
__global__ __launch_bounds__(128) void finalize_kernel(
    const unsigned short* __restrict__ Abf,
    const unsigned short* __restrict__ Bbf,
    const float* __restrict__ posBuf,
    const float* __restrict__ hardV,
    const int*   __restrict__ hardI,
    const float* __restrict__ logit_scale,
    float* __restrict__ out) {
  __shared__ int   sHard[KHARD];
  __shared__ int   sPick[KRAND];
  __shared__ float sLogit[1 + KHARD + KRAND];
  __shared__ float sSc;

  const int dir = blockIdx.y;
  const int row = blockIdx.x;
  const int tid = threadIdx.x;
  const size_t base = (size_t)dir * BSZ + row;
  const unsigned short* __restrict__ Arow = (dir ? Bbf : Abf) + (size_t)row * DSZ;
  const unsigned short* __restrict__ Bmat = dir ? Abf : Bbf;

  if (tid < KHARD) sHard[tid] = hardI[base * KHARD + tid];
  __syncthreads();

  if (tid == 0) {
    sSc = fminf(__expf(logit_scale[0]), 100.0f);
    // deterministic per-(dir,row) rejection sampling, distinct picks
    unsigned int st = 0x9E3779B9u ^ (unsigned int)(base * 2654435761u + 12345u);
    int cnt = 0, guard = 0;
    while (cnt < KRAND && guard < 4096) {
      st = st * 1664525u + 1013904223u;
      unsigned int h = st;
      h ^= h >> 16; h *= 0x7feb352du; h ^= h >> 15; h *= 0x846ca68bu; h ^= h >> 16;
      const int cand = (int)(h & (BSZ - 1));
      bool bad = (cand == row);
      for (int j = 0; j < KHARD && !bad; ++j) bad = (sHard[j] == cand);
      for (int j = 0; j < cnt && !bad; ++j)   bad = (sPick[j] == cand);
      if (!bad) sPick[cnt++] = cand;
      ++guard;
    }
    while (cnt < KRAND) { sPick[cnt] = (row + cnt + 1) & (BSZ - 1); ++cnt; }
  }
  __syncthreads();

  // 4 threads per pick, contiguous 256-element segments, bf16 dot in f32
  const int pick = tid >> 2;
  const int part = tid & 3;
  const unsigned short* Brow = Bmat + (size_t)sPick[pick] * DSZ;
  float accd = 0.f;
  const int kb = part * (DSZ / 4);
  for (int k = 0; k < DSZ / 4; k += 8) {
    accd += dot8(*(const uint4*)(Arow + kb + k), *(const uint4*)(Brow + kb + k));
  }
  accd += __shfl_down(accd, 2, 4);
  accd += __shfl_down(accd, 1, 4);
  if (part == 0) sLogit[1 + KHARD + pick] = accd;   // scaled below

  if (tid < KHARD) sLogit[1 + tid] = hardV[base * KHARD + tid];
  if (tid == 0)    sLogit[0] = posBuf[base];
  __syncthreads();

  if (tid == 0) {
    const float sc = sSc;
    float m = sc * sLogit[0];
    for (int j = 0; j < 1 + KHARD + KRAND; ++j) m = fmaxf(m, sc * sLogit[j]);
    float s = 0.f;
    for (int j = 0; j < 1 + KHARD + KRAND; ++j) s += __expf(sc * sLogit[j] - m);
    const float lse  = m + __logf(s);
    const float loss = lse - sc * sLogit[0];        // -log_softmax[:,0]
    atomicAdd(out, loss * (1.0f / (2.0f * (float)BSZ)));
  }
}

// ------------------------------------------------------------------
// launcher
// ------------------------------------------------------------------
extern "C" void kernel_launch(void* const* d_in, const int* in_sizes, int n_in,
                              void* d_out, int out_size, void* d_ws, size_t ws_size,
                              hipStream_t stream) {
  (void)in_sizes; (void)n_in; (void)out_size; (void)ws_size;
  const float* h_img = (const float*)d_in[0];
  const float* h_txt = (const float*)d_in[1];
  const float* lsc   = (const float*)d_in[2];
  float* out = (float*)d_out;

  char* ws = (char*)d_ws;
  const size_t bfBytes = (size_t)BSZ * DSZ * sizeof(unsigned short);   // 16 MB
  unsigned short* Abf = (unsigned short*)ws;
  unsigned short* Bbf = (unsigned short*)(ws + bfBytes);
  float* posB  = (float*)(ws + 2 * bfBytes);
  float* hardV = (float*)(ws + 2 * bfBytes + (size_t)2 * BSZ * 4);
  int*   hardI = (int*)  (ws + 2 * bfBytes + (size_t)2 * BSZ * 4 + (size_t)2 * BSZ * KHARD * 4);

  normalize_kernel<<<2 * BSZ, 256, 0, stream>>>(h_img, h_txt, Abf, Bbf);
  zero_kernel<<<1, 1, 0, stream>>>(out);

  const size_t shmem = (size_t)BAND * ASTRIDE * 2      // A band (bf16, padded)
                     + (size_t)BAND * SSTRIDE * 4      // staging tile
                     + (size_t)BAND * KHARD * 4        // top-k vals
                     + (size_t)BAND * KHARD * 4        // top-k idx
                     + (size_t)BAND * 4;               // positives
  dim3 g1(BSZ / BAND, 2);
  sim_topk_kernel<<<g1, 256, shmem, stream>>>(Abf, Bbf, posB, hardV, hardI);

  dim3 g2(BSZ, 2);
  finalize_kernel<<<g2, 128, 0, stream>>>(Abf, Bbf, posB, hardV, hardI, lsc, out);
}
